// PointNetgaussSsm_61546881352082
// MI455X (gfx1250) — matrix operand order repaired and checked by reading, not compiled, not run on hardware
//
#include <hip/hip_runtime.h>
#include <hip/hip_bf16.h>

typedef __attribute__((ext_vector_type(16))) _Float16 v16h;
typedef __attribute__((ext_vector_type(8)))  _Float16 v8h;
typedef __attribute__((ext_vector_type(8)))  float    v8f;

#define NEG_INF (-3.0e38f)
#define POS_INF (3.0e38f)

// ---------------------------------------------------------------------------
// WMMA helpers (gfx1250 wave32, v_wmma_f32_16x16x32_f16)
// A fragment: 16x32 f16 row-major (lane m = lane&15; K-halves split by lane>>4)
//   -> two contiguous 16-byte loads per lane.
// B fragment: from pre-transposed f16 weights Wt[N][KPAD] (lane n = lane&15,
//   k-block by lane>>4) -> two contiguous 16-byte loads per lane.
// C/D: 8 f32 per lane; VGPR i -> row i + 8*(lane>>4), col lane&15
// ---------------------------------------------------------------------------
__device__ inline v16h load_a_frag(const _Float16* A, int lda) {
  const int lane = threadIdx.x & 31;
  const int m  = lane & 15;
  const int kb = (lane >> 4) << 3;  // 0 or 8
  const _Float16* p = A + m * lda + kb;
  v8h lo = *(const v8h*)(p);
  v8h hi = *(const v8h*)(p + 16);
  return __builtin_shufflevector(lo, hi, 0, 1, 2, 3, 4, 5, 6, 7,
                                 8, 9, 10, 11, 12, 13, 14, 15);
}

__device__ inline v16h load_b_frag_t(const _Float16* __restrict__ Wt, int ldk,
                                     int kbase, int ncol0) {
  const int lane = threadIdx.x & 31;
  const int n  = ncol0 + (lane & 15);
  const int k0 = kbase + ((lane >> 4) << 3);  // 0 or 8
  const _Float16* p = Wt + (size_t)n * ldk + k0;
  v8h lo = *(const v8h*)(p);
  v8h hi = *(const v8h*)(p + 16);
  return __builtin_shufflevector(lo, hi, 0, 1, 2, 3, 4, 5, 6, 7,
                                 8, 9, 10, 11, 12, 13, 14, 15);
}

__device__ inline v8f wmma_f16f32(v16h a, v16h b, v8f c) {
  return __builtin_amdgcn_wmma_f32_16x16x32_f16(false, a, false, b, (short)0, c,
                                                false, false);
}

// GEMM layer: [ROWS x KPAD(lds,f16)] x Wt[NOUT x KPAD](global,f16,transposed)
// + bias, y = y*g + b2, relu, store f16 to LDS. blockDim must be 128 (4 waves).
template <int ROWS, int KPAD, int NOUT>
__device__ inline void gemm_bn_relu(const _Float16* A, int lda,
                                    const _Float16* __restrict__ Wt,
                                    const float* __restrict__ bias,
                                    const float* __restrict__ gam,
                                    const float* __restrict__ bet,
                                    _Float16* O, int ldo) {
  const int wave = threadIdx.x >> 5;
  const int lane = threadIdx.x & 31;
  constexpr int RT = ROWS / 16;
  constexpr int CT = NOUT / 16;
  for (int t = wave; t < RT * CT; t += 4) {
    const int rt = t / CT, ct = t % CT;
    v8f acc = {};
#pragma unroll 2
    for (int kk = 0; kk < KPAD; kk += 32) {
      v16h a = load_a_frag(A + rt * 16 * lda + kk, lda);
      v16h b = load_b_frag_t(Wt, KPAD, kk, ct * 16);
      acc = wmma_f16f32(a, b, acc);
    }
    const int col = ct * 16 + (lane & 15);
    const float bs = bias[col];
    const float gm = gam[col];
    const float bt = bet[col];
#pragma unroll
    for (int i = 0; i < 8; ++i) {
      const int row = rt * 16 + i + 8 * (lane >> 4);
      float v = acc[i] + bs;
      v = v * gm + bt;
      v = v > 0.0f ? v : 0.0f;
      O[row * ldo + col] = (_Float16)v;
    }
  }
}

// ---------------------------------------------------------------------------
// Weight prep: W[K][N] f32 row-major -> Wt[N][KPAD] f16, zero-padded K.
// ---------------------------------------------------------------------------
__global__ void __launch_bounds__(256)
convert_wt_kernel(const float* __restrict__ W, int K, int N, int KPAD,
                  _Float16* __restrict__ Wt) {
  const int idx = blockIdx.x * 256 + threadIdx.x;
  if (idx >= N * KPAD) return;
  const int n = idx / KPAD, k = idx % KPAD;
  const float v = (k < K) ? W[(size_t)k * N + n] : 0.0f;
  Wt[idx] = (_Float16)v;
}

// ---------------------------------------------------------------------------
// FPS: one block per batch; sequential farthest point sampling.
// ---------------------------------------------------------------------------
__global__ void __launch_bounds__(256)
fps_kernel(const float* __restrict__ pos, int N, int S, float* __restrict__ pos_out) {
  __shared__ float d[4096];
  __shared__ float redv[256];
  __shared__ int   redi[256];
  __shared__ float curp[3];
  const int b = blockIdx.x, tid = threadIdx.x;
  const float* p  = pos + (size_t)b * N * 3;
  float*       po = pos_out + (size_t)b * S * 3;
  if (tid < 3) { curp[tid] = p[tid]; po[tid] = p[tid]; }
  __syncthreads();
  for (int j = tid; j < N; j += 256) {
    float dx = p[3 * j] - curp[0], dy = p[3 * j + 1] - curp[1], dz = p[3 * j + 2] - curp[2];
    d[j] = dx * dx + dy * dy + dz * dz;
  }
  __syncthreads();
  for (int s = 1; s < S; ++s) {
    float lv = -1.0f; int li = 0;
    for (int j = tid; j < N; j += 256) {
      if (d[j] > lv) { lv = d[j]; li = j; }
    }
    redv[tid] = lv; redi[tid] = li;
    __syncthreads();
    for (int off = 128; off > 0; off >>= 1) {
      if (tid < off) {
        float ov = redv[tid + off]; int oi = redi[tid + off];
        if (ov > redv[tid] || (ov == redv[tid] && oi < redi[tid])) {
          redv[tid] = ov; redi[tid] = oi;
        }
      }
      __syncthreads();
    }
    const int bi = redi[0];
    if (tid < 3) { curp[tid] = p[3 * bi + tid]; po[3 * s + tid] = p[3 * bi + tid]; }
    __syncthreads();
    for (int j = tid; j < N; j += 256) {
      float dx = p[3 * j] - curp[0], dy = p[3 * j + 1] - curp[1], dz = p[3 * j + 2] - curp[2];
      float dd = dx * dx + dy * dy + dz * dz;
      if (dd < d[j]) d[j] = dd;
    }
    __syncthreads();
  }
}

// ---------------------------------------------------------------------------
// SA module: per query, select 64 nearest neighbors, build [64 x (CX+3)] feats,
// run MLP (CIN->H1->H2->COUT) with WMMA, masked max-pool over neighbors.
// grid = (M, B), block = 128. Weights pre-transposed f16.
// ---------------------------------------------------------------------------
template <int NSRC, int CX, int KP0, int H1, int H2, int COUT, int APITCH>
__global__ void __launch_bounds__(128)
sa_kernel(const float* __restrict__ pos_src, const float* __restrict__ x_src,
          const float* __restrict__ pos_q, int M,
          const _Float16* __restrict__ Wt0, const float* __restrict__ b0,
          const float* __restrict__ g0, const float* __restrict__ bb0,
          const _Float16* __restrict__ Wt1, const float* __restrict__ b1,
          const float* __restrict__ g1, const float* __restrict__ bb1,
          const _Float16* __restrict__ Wt2, const float* __restrict__ b2c,
          float r2, float* __restrict__ x_out) {
  constexpr int CIN = CX + 3;
  __shared__ float    d2s[NSRC];
  __shared__ float    redv[128];
  __shared__ int      redi[128];
  __shared__ int      nidx[64];
  __shared__ float    ndist[64];
  __shared__ alignas(16) _Float16 bufA[64 * APITCH];
  __shared__ alignas(16) _Float16 bufB[64 * H1];
  __shared__ float    partial[4][COUT];

  const int q = blockIdx.x, b = blockIdx.y, tid = threadIdx.x;
  const float* ps = pos_src + (size_t)b * NSRC * 3;
  const float* xs = x_src + (size_t)b * NSRC * CX;
  const float* pq = pos_q + ((size_t)b * M + q) * 3;
  const float qx = pq[0], qy = pq[1], qz = pq[2];

  for (int j = tid; j < NSRC; j += 128) {
    float dx = ps[3 * j] - qx, dy = ps[3 * j + 1] - qy, dz = ps[3 * j + 2] - qz;
    d2s[j] = dx * dx + dy * dy + dz * dz;
  }
  __syncthreads();

  // iterative K=64 smallest selection
  for (int s = 0; s < 64; ++s) {
    float lv = POS_INF; int li = 0;
    for (int j = tid; j < NSRC; j += 128) {
      float v = d2s[j];
      if (v < lv) { lv = v; li = j; }
    }
    redv[tid] = lv; redi[tid] = li;
    __syncthreads();
    for (int off = 64; off > 0; off >>= 1) {
      if (tid < off) {
        float ov = redv[tid + off]; int oi = redi[tid + off];
        if (ov < redv[tid] || (ov == redv[tid] && oi < redi[tid])) {
          redv[tid] = ov; redi[tid] = oi;
        }
      }
      __syncthreads();
    }
    if (tid == 0) {
      nidx[s] = redi[0];
      ndist[s] = redv[0];
      d2s[redi[0]] = POS_INF;
    }
    __syncthreads();
  }

  // build feats [64 x KP0] (zero padded): [x_src[j], pos_src[j]-q]
  for (int e = tid; e < 64 * KP0; e += 128) {
    const int k = e / KP0, c = e % KP0;
    const int j = nidx[k];
    float v = 0.0f;
    if (c < CX) {
      v = xs[(size_t)j * CX + c];
    } else if (c < CIN) {
      const int dd = c - CX;
      v = ps[3 * j + dd] - (dd == 0 ? qx : (dd == 1 ? qy : qz));
    }
    bufA[k * KP0 + c] = (_Float16)v;
  }
  __syncthreads();

  gemm_bn_relu<64, KP0, H1>(bufA, KP0, Wt0, b0, g0, bb0, bufB, H1);
  __syncthreads();
  gemm_bn_relu<64, H1, H2>(bufB, H1, Wt1, b1, g1, bb1, bufA, H2);
  __syncthreads();

  // last layer (plain) fused with masked max over the 64 neighbor rows
  const int wave = tid >> 5, lane = tid & 31;
  constexpr int CT = COUT / 16;
  for (int t = wave; t < 4 * CT; t += 4) {
    const int rt = t / CT, ct = t % CT;
    v8f acc = {};
#pragma unroll 2
    for (int kk = 0; kk < H2; kk += 32) {
      v16h a = load_a_frag(bufA + rt * 16 * H2 + kk, H2);
      v16h bf = load_b_frag_t(Wt2, H2, kk, ct * 16);
      acc = wmma_f16f32(a, bf, acc);
    }
    const int col = ct * 16 + (lane & 15);
    const float bs = b2c[col];
    float mx = NEG_INF;
#pragma unroll
    for (int i = 0; i < 8; ++i) {
      const int row = rt * 16 + i + 8 * (lane >> 4);
      const bool valid = ndist[row] <= r2;
      const float v = acc[i] + bs;
      if (valid && v > mx) mx = v;
    }
    const float o = __shfl_xor(mx, 16, 32);
    mx = mx > o ? mx : o;
    if (lane < 16) partial[rt][col] = mx;
  }
  __syncthreads();
  for (int c = tid; c < COUT; c += 128) {
    float m = partial[0][c];
#pragma unroll
    for (int rt = 1; rt < 4; ++rt) m = fmaxf(m, partial[rt][c]);
    x_out[((size_t)b * M + q) * COUT + c] = m;
  }
}

// ---------------------------------------------------------------------------
// TransitionDown per-point MLP: Y = relu(bn(X @ W + b)), WMMA GEMM.
// 64 rows per block, block = 128.
// ---------------------------------------------------------------------------
template <int KDIM, int NOUT>
__global__ void __launch_bounds__(128)
mlp_bn_relu_kernel(const float* __restrict__ X, const _Float16* __restrict__ Wt,
                   const float* __restrict__ bias, const float* __restrict__ gam,
                   const float* __restrict__ bet, float* __restrict__ Y, int nrows) {
  __shared__ alignas(16) _Float16 bufA[64 * KDIM];
  const int r0 = blockIdx.x * 64;
  const int tid = threadIdx.x;
  if (r0 >= nrows) return;
  for (int e = tid; e < 64 * KDIM; e += 128)
    bufA[e] = (_Float16)X[(size_t)r0 * KDIM + e];
  __syncthreads();
  const int wave = tid >> 5, lane = tid & 31;
  constexpr int CT = NOUT / 16;
  for (int t = wave; t < 4 * CT; t += 4) {
    const int rt = t / CT, ct = t % CT;
    v8f acc = {};
#pragma unroll 2
    for (int kk = 0; kk < KDIM; kk += 32) {
      v16h a = load_a_frag(bufA + rt * 16 * KDIM + kk, KDIM);
      v16h b = load_b_frag_t(Wt, KDIM, kk, ct * 16);
      acc = wmma_f16f32(a, b, acc);
    }
    const int col = ct * 16 + (lane & 15);
    const float bs = bias[col], gm = gam[col], bt = bet[col];
#pragma unroll
    for (int i = 0; i < 8; ++i) {
      const int row = r0 + rt * 16 + i + 8 * (lane >> 4);
      float v = acc[i] + bs;
      v = v * gm + bt;
      v = v > 0.0f ? v : 0.0f;
      Y[(size_t)row * NOUT + col] = v;
    }
  }
}

// ---------------------------------------------------------------------------
// knn(k=KN) + feature max ("scatter-max"). grid = (M, B), block = 128.
// ---------------------------------------------------------------------------
template <int NSRC, int CFEAT, int KN>
__global__ void __launch_bounds__(128)
knn_max_kernel(const float* __restrict__ pos_src, const float* __restrict__ pos_q,
               const float* __restrict__ feat_src, int M, float* __restrict__ out) {
  __shared__ float d2s[NSRC];
  __shared__ float redv[128];
  __shared__ int   redi[128];
  __shared__ int   nidx[KN];
  const int q = blockIdx.x, b = blockIdx.y, tid = threadIdx.x;
  const float* ps = pos_src + (size_t)b * NSRC * 3;
  const float* pq = pos_q + ((size_t)b * M + q) * 3;
  const float qx = pq[0], qy = pq[1], qz = pq[2];
  for (int j = tid; j < NSRC; j += 128) {
    float dx = ps[3 * j] - qx, dy = ps[3 * j + 1] - qy, dz = ps[3 * j + 2] - qz;
    d2s[j] = dx * dx + dy * dy + dz * dz;
  }
  __syncthreads();
  for (int s = 0; s < KN; ++s) {
    float lv = POS_INF; int li = 0;
    for (int j = tid; j < NSRC; j += 128) {
      float v = d2s[j];
      if (v < lv) { lv = v; li = j; }
    }
    redv[tid] = lv; redi[tid] = li;
    __syncthreads();
    for (int off = 64; off > 0; off >>= 1) {
      if (tid < off) {
        float ov = redv[tid + off]; int oi = redi[tid + off];
        if (ov < redv[tid] || (ov == redv[tid] && oi < redi[tid])) {
          redv[tid] = ov; redi[tid] = oi;
        }
      }
      __syncthreads();
    }
    if (tid == 0) { nidx[s] = redi[0]; d2s[redi[0]] = POS_INF; }
    __syncthreads();
  }
  const float* fs = feat_src + (size_t)b * NSRC * CFEAT;
  for (int c = tid; c < CFEAT; c += 128) {
    float m = NEG_INF;
#pragma unroll 4
    for (int k = 0; k < KN; ++k) {
      float v = fs[(size_t)nidx[k] * CFEAT + c];
      if (v > m) m = v;
    }
    out[((size_t)b * M + q) * CFEAT + c] = m;
  }
}

// ---------------------------------------------------------------------------
// GlobalSAModule: feats=[x2d|pos2d] (32 x 515) -> 256 -> 512 -> 1024, mean pool.
// grid = B, block = 128.
// ---------------------------------------------------------------------------
__global__ void __launch_bounds__(128)
gsa_kernel(const float* __restrict__ x_src, const float* __restrict__ pos_src,
           const _Float16* __restrict__ Wt0, const float* __restrict__ b0,
           const float* __restrict__ g0, const float* __restrict__ bb0,
           const _Float16* __restrict__ Wt1, const float* __restrict__ b1,
           const float* __restrict__ g1, const float* __restrict__ bb1,
           const _Float16* __restrict__ Wt2, const float* __restrict__ b2c,
           float* __restrict__ g_out) {
  constexpr int ROWS = 32, CIN = 515, KP0 = 544, H1 = 256, H2 = 512, COUT = 1024;
  __shared__ alignas(16) _Float16 bufA[ROWS * KP0];  // also holds h2 (ROWS*H2 <= ROWS*KP0)
  __shared__ alignas(16) _Float16 bufB[ROWS * H1];
  __shared__ float    partial[2][COUT];
  const int b = blockIdx.x, tid = threadIdx.x;
  for (int e = tid; e < ROWS * KP0; e += 128) {
    const int r = e / KP0, c = e % KP0;
    float v = 0.0f;
    if (c < 512)      v = x_src[((size_t)b * ROWS + r) * 512 + c];
    else if (c < CIN) v = pos_src[((size_t)b * ROWS + r) * 3 + (c - 512)];
    bufA[e] = (_Float16)v;
  }
  __syncthreads();
  gemm_bn_relu<ROWS, KP0, H1>(bufA, KP0, Wt0, b0, g0, bb0, bufB, H1);
  __syncthreads();
  gemm_bn_relu<ROWS, H1, H2>(bufB, H1, Wt1, b1, g1, bb1, bufA, H2);
  __syncthreads();
  // last layer + mean over 32 rows
  const int wave = tid >> 5, lane = tid & 31;
  constexpr int CT = COUT / 16;
  for (int t = wave; t < 2 * CT; t += 4) {
    const int rt = t / CT, ct = t % CT;
    v8f acc = {};
#pragma unroll 2
    for (int kk = 0; kk < H2; kk += 32) {
      v16h a = load_a_frag(bufA + rt * 16 * H2 + kk, H2);
      v16h bf = load_b_frag_t(Wt2, H2, kk, ct * 16);
      acc = wmma_f16f32(a, bf, acc);
    }
    const int col = ct * 16 + (lane & 15);
    const float bs = b2c[col];
    float sum = 0.0f;
#pragma unroll
    for (int i = 0; i < 8; ++i) sum += acc[i] + bs;
    sum += __shfl_xor(sum, 16, 32);
    if (lane < 16) partial[rt][col] = sum;
  }
  __syncthreads();
  for (int c = tid; c < COUT; c += 128)
    g_out[(size_t)b * COUT + c] = (partial[0][c] + partial[1][c]) * (1.0f / 32.0f);
}

// ---------------------------------------------------------------------------
// Classification head: 4 rows, 1024->512->256->40 (relu, relu, plain).
// ---------------------------------------------------------------------------
__global__ void __launch_bounds__(256)
head_kernel(const float* __restrict__ g,
            const float* __restrict__ W0, const float* __restrict__ b0,
            const float* __restrict__ W1, const float* __restrict__ b1,
            const float* __restrict__ W2, const float* __restrict__ b2,
            float* __restrict__ out) {
  __shared__ float t1[4 * 512];
  __shared__ float t2[4 * 256];
  const int tid = threadIdx.x;
  for (int e = tid; e < 4 * 512; e += 256) {
    const int r = e / 512, c = e % 512;
    float s = b0[c];
    for (int k = 0; k < 1024; ++k) s += g[r * 1024 + k] * W0[k * 512 + c];
    t1[e] = s > 0.0f ? s : 0.0f;
  }
  __syncthreads();
  for (int e = tid; e < 4 * 256; e += 256) {
    const int r = e / 256, c = e % 256;
    float s = b1[c];
    for (int k = 0; k < 512; ++k) s += t1[r * 512 + k] * W1[k * 256 + c];
    t2[e] = s > 0.0f ? s : 0.0f;
  }
  __syncthreads();
  for (int e = tid; e < 4 * 40; e += 256) {
    const int r = e / 40, c = e % 40;
    float s = b2[c];
    for (int k = 0; k < 256; ++k) s += t2[r * 256 + k] * W2[k * 40 + c];
    out[e] = s;
  }
}

// ---------------------------------------------------------------------------
extern "C" void kernel_launch(void* const* d_in, const int* in_sizes, int n_in,
                              void* d_out, int out_size, void* d_ws, size_t ws_size,
                              hipStream_t stream) {
  (void)in_sizes; (void)n_in; (void)out_size; (void)ws_size;

  int i = 0;
  const float* data = (const float*)d_in[i++];  // [4,4096,3]
  // sa1: [6,64,64,128]
  const float *sa1W0 = (const float*)d_in[i++], *sa1b0 = (const float*)d_in[i++],
              *sa1g0 = (const float*)d_in[i++], *sa1bb0 = (const float*)d_in[i++];
  const float *sa1W1 = (const float*)d_in[i++], *sa1b1 = (const float*)d_in[i++],
              *sa1g1 = (const float*)d_in[i++], *sa1bb1 = (const float*)d_in[i++];
  const float *sa1W2 = (const float*)d_in[i++], *sa1b2 = (const float*)d_in[i++];
  // td1: [128,256]
  const float *td1W = (const float*)d_in[i++], *td1b = (const float*)d_in[i++],
              *td1g = (const float*)d_in[i++], *td1bb = (const float*)d_in[i++];
  // sa2: [259,128,128,256]
  const float *sa2W0 = (const float*)d_in[i++], *sa2b0 = (const float*)d_in[i++],
              *sa2g0 = (const float*)d_in[i++], *sa2bb0 = (const float*)d_in[i++];
  const float *sa2W1 = (const float*)d_in[i++], *sa2b1 = (const float*)d_in[i++],
              *sa2g1 = (const float*)d_in[i++], *sa2bb1 = (const float*)d_in[i++];
  const float *sa2W2 = (const float*)d_in[i++], *sa2b2 = (const float*)d_in[i++];
  // td2: [256,512]
  const float *td2W = (const float*)d_in[i++], *td2b = (const float*)d_in[i++],
              *td2g = (const float*)d_in[i++], *td2bb = (const float*)d_in[i++];
  // sa3: [515,256,512,1024]
  const float *sa3W0 = (const float*)d_in[i++], *sa3b0 = (const float*)d_in[i++],
              *sa3g0 = (const float*)d_in[i++], *sa3bb0 = (const float*)d_in[i++];
  const float *sa3W1 = (const float*)d_in[i++], *sa3b1 = (const float*)d_in[i++],
              *sa3g1 = (const float*)d_in[i++], *sa3bb1 = (const float*)d_in[i++];
  const float *sa3W2 = (const float*)d_in[i++], *sa3b2 = (const float*)d_in[i++];
  // head: [1024,512,256,40]
  const float *hW0 = (const float*)d_in[i++], *hb0 = (const float*)d_in[i++];
  const float *hW1 = (const float*)d_in[i++], *hb1 = (const float*)d_in[i++];
  const float *hW2 = (const float*)d_in[i++], *hb2 = (const float*)d_in[i++];

  // workspace layout (float units; every chunk is a multiple of 8 floats -> 32B aligned)
  float* ws = (float*)d_ws;
  size_t off = 0;
  auto alloc = [&](size_t n) { float* p = ws + off; off += n; return p; };
  float* pos1  = alloc((size_t)4 * 2048 * 3);
  float* pos1d = alloc((size_t)4 * 512 * 3);
  float* pos2  = alloc((size_t)4 * 128 * 3);
  float* pos2d = alloc((size_t)4 * 32 * 3);
  float* x1    = alloc((size_t)4 * 2048 * 128);
  float* h1    = alloc((size_t)4 * 2048 * 256);
  float* x1d   = alloc((size_t)4 * 512 * 256);
  float* x2    = alloc((size_t)4 * 128 * 256);
  float* h2    = alloc((size_t)4 * 128 * 512);
  float* x2d   = alloc((size_t)4 * 32 * 512);
  float* gbuf  = alloc((size_t)4 * 1024);
  // f16 transposed/padded weights
  auto alloch = [&](size_t nh) { _Float16* p = (_Float16*)(ws + off); off += nh / 2; return p; };
  _Float16* t_sa1W0 = alloch((size_t)64 * 32);
  _Float16* t_sa1W1 = alloch((size_t)64 * 64);
  _Float16* t_sa1W2 = alloch((size_t)128 * 64);
  _Float16* t_td1W  = alloch((size_t)256 * 128);
  _Float16* t_sa2W0 = alloch((size_t)128 * 288);
  _Float16* t_sa2W1 = alloch((size_t)128 * 128);
  _Float16* t_sa2W2 = alloch((size_t)256 * 128);
  _Float16* t_td2W  = alloch((size_t)512 * 256);
  _Float16* t_sa3W0 = alloch((size_t)256 * 544);
  _Float16* t_sa3W1 = alloch((size_t)512 * 256);
  _Float16* t_sa3W2 = alloch((size_t)1024 * 512);

  auto cvt = [&](const float* W, int K, int N, int KPAD, _Float16* Wt) {
    const int total = N * KPAD;
    convert_wt_kernel<<<(total + 255) / 256, 256, 0, stream>>>(W, K, N, KPAD, Wt);
  };
  cvt(sa1W0, 6, 64, 32, t_sa1W0);
  cvt(sa1W1, 64, 64, 64, t_sa1W1);
  cvt(sa1W2, 64, 128, 64, t_sa1W2);
  cvt(td1W, 128, 256, 128, t_td1W);
  cvt(sa2W0, 259, 128, 288, t_sa2W0);
  cvt(sa2W1, 128, 128, 128, t_sa2W1);
  cvt(sa2W2, 128, 256, 128, t_sa2W2);
  cvt(td2W, 256, 512, 256, t_td2W);
  cvt(sa3W0, 515, 256, 544, t_sa3W0);
  cvt(sa3W1, 256, 512, 256, t_sa3W1);
  cvt(sa3W2, 512, 1024, 512, t_sa3W2);

  // SA1: fps 4096->2048, r=0.2, K=64, MLP 6->64->64->128
  fps_kernel<<<4, 256, 0, stream>>>(data, 4096, 2048, pos1);
  sa_kernel<4096, 3, 32, 64, 64, 128, 64><<<dim3(2048, 4), 128, 0, stream>>>(
      data, data, pos1, 2048,
      t_sa1W0, sa1b0, sa1g0, sa1bb0, t_sa1W1, sa1b1, sa1g1, sa1bb1, t_sa1W2, sa1b2,
      0.2f * 0.2f, x1);

  // TD1: fps 2048->512, MLP 128->256, knn16 max
  fps_kernel<<<4, 256, 0, stream>>>(pos1, 2048, 512, pos1d);
  mlp_bn_relu_kernel<128, 256><<<(4 * 2048) / 64, 128, 0, stream>>>(
      x1, t_td1W, td1b, td1g, td1bb, h1, 4 * 2048);
  knn_max_kernel<2048, 256, 16><<<dim3(512, 4), 128, 0, stream>>>(
      pos1, pos1d, h1, 512, x1d);

  // SA2: fps 512->128, r=0.4, K=64, MLP 259->128->128->256
  fps_kernel<<<4, 256, 0, stream>>>(pos1d, 512, 128, pos2);
  sa_kernel<512, 256, 288, 128, 128, 256, 288><<<dim3(128, 4), 128, 0, stream>>>(
      pos1d, x1d, pos2, 128,
      t_sa2W0, sa2b0, sa2g0, sa2bb0, t_sa2W1, sa2b1, sa2g1, sa2bb1, t_sa2W2, sa2b2,
      0.4f * 0.4f, x2);

  // TD2: fps 128->32, MLP 256->512, knn16 max
  fps_kernel<<<4, 256, 0, stream>>>(pos2, 128, 32, pos2d);
  mlp_bn_relu_kernel<256, 512><<<(4 * 128) / 64, 128, 0, stream>>>(
      x2, t_td2W, td2b, td2g, td2bb, h2, 4 * 128);
  knn_max_kernel<128, 512, 16><<<dim3(32, 4), 128, 0, stream>>>(
      pos2, pos2d, h2, 32, x2d);

  // GlobalSA: MLP 515->256->512->1024, mean pool
  gsa_kernel<<<4, 128, 0, stream>>>(x2d, pos2d,
      t_sa3W0, sa3b0, sa3g0, sa3bb0, t_sa3W1, sa3b1, sa3g1, sa3bb1, t_sa3W2, sa3b2,
      gbuf);

  // head: 1024->512->256->40
  head_kernel<<<1, 256, 0, stream>>>(gbuf, hW0, hb0, hW1, hb1, hW2, hb2,
                                     (float*)d_out);
}
